// AgnisV5_5626407158378
// MI455X (gfx1250) — matrix operand order repaired and checked by reading, not compiled, not run on hardware
//
#include <hip/hip_runtime.h>
#include <hip/hip_bf16.h>

typedef __bf16 bf16_t;
typedef __attribute__((ext_vector_type(16))) __bf16 v16bf;
typedef __attribute__((ext_vector_type(8)))  __bf16 v8bf;
typedef __attribute__((ext_vector_type(8)))  float  v8f;

#define D_MODEL  768
#define H_DIM    3072
#define SEQ_T    256
#define BATCH    4
#define VOCAB    50257
#define VOCAB_PAD 50304   // 3144 tiles of 16 -> 786 groups of 4 tiles
#define ALPHA    0.4f

// ---------------------------------------------------------------------------
// helpers
// ---------------------------------------------------------------------------
__device__ __forceinline__ float gelu_exact(float x) {
  return 0.5f * x * (1.0f + erff(x * 0.70710678118654752f));
}

__device__ __forceinline__ float wave_sum(float v) {
  #pragma unroll
  for (int m = 16; m >= 1; m >>= 1) v += __shfl_xor(v, m, 32);
  return v;
}

// One wave computes NB adjacent 16x16 output tiles (N-register-blocking):
//   D[j] = A(16xK) * W^T rows [nb0+16j, nb0+16j+16),  j = 0..NB-1
// Per K-step: ONE A-fragment (consumed immediately -> no LICM hoist/spill)
// and NB contiguous-32B B-fragments, then NB back-to-back WMMAs.
template<int NB>
__device__ __forceinline__ void tile_gemm_nb(const bf16_t* A, int lda,
                                             const bf16_t* WT, int K,
                                             int nb0, int lane, v8f* acc) {
  const int m  = lane & 15;
  const int hi = lane >> 4;
  const bf16_t* ap = A  + (size_t)m * lda + hi * 16;
  const bf16_t* bp = WT + (size_t)(nb0 + m) * K + hi * 16;
  #pragma unroll
  for (int j = 0; j < NB; ++j) acc[j] = v8f{};
  #pragma clang loop unroll_count(2)
  for (int kb = 0; kb < K; kb += 32) {
    v8bf alo = *(const v8bf*)(ap + kb);
    v8bf ahi = *(const v8bf*)(ap + kb + 8);
    v16bf a;
    #pragma unroll
    for (int i = 0; i < 8; ++i) { a[i] = alo[i]; a[i + 8] = ahi[i]; }
    #pragma unroll
    for (int j = 0; j < NB; ++j) {
      v16bf b = *(const v16bf*)(bp + (size_t)j * 16 * K + kb);
      acc[j] = __builtin_amdgcn_wmma_f32_16x16x32_bf16(
          false, a, false, b, (short)0, acc[j], false, false);
    }
  }
}

// ---------------------------------------------------------------------------
// one-time: f32 [K][N] weight  ->  bf16 W^T [Npad][K]  (zero-pad extra rows)
// ---------------------------------------------------------------------------
__global__ void convert_transpose(const float* __restrict__ src,
                                  bf16_t* __restrict__ dst,
                                  int K, int N, int Npad) {
  size_t idx = (size_t)blockIdx.x * blockDim.x + threadIdx.x;
  size_t total = (size_t)Npad * K;
  if (idx >= total) return;
  int k = (int)(idx % (size_t)K);
  int n = (int)(idx / (size_t)K);
  float v = (n < N) ? src[(size_t)k * N + n] : 0.0f;
  dst[idx] = (bf16_t)v;
}

__global__ void init_state(bf16_t* __restrict__ h, float* __restrict__ x2) {
  int i = blockIdx.x * blockDim.x + threadIdx.x;
  if (i < 16 * D_MODEL) h[i] = (bf16_t)0.0f;
  if (i < BATCH * D_MODEL) x2[i] = 0.0f;
}

// ---------------------------------------------------------------------------
// per-timestep "small model" step: one 16-wave workgroup, LDS-staged.
// 512 threads (4 waves/SIMD) keeps the per-wave VGPR budget large -> no spill.
// Updates h_state (bf16 [16][768], rows 0..3 valid) and x2_state (f32 [4][768]).
// ---------------------------------------------------------------------------
__global__ void __launch_bounds__(512)
step_kernel(const int* __restrict__ token_ids, const float* __restrict__ embedding,
            const float* __restrict__ b0v, const float* __restrict__ b1v,
            const float* __restrict__ c1v, const float* __restrict__ c2v,
            const float* __restrict__ bgv, const float* __restrict__ gamv,
            const float* __restrict__ betv,
            const bf16_t* __restrict__ V0T, const bf16_t* __restrict__ V1T,
            const bf16_t* __restrict__ W1T, const bf16_t* __restrict__ W2T,
            const bf16_t* __restrict__ WgT, const bf16_t* __restrict__ WtT,
            const bf16_t* __restrict__ RT,
            bf16_t* __restrict__ h_state, float* __restrict__ x2_state,
            int t) {
  extern __shared__ char smem[];
  float*  s_embN   = (float*)(smem);                 // [4][768]
  float*  s_traw   = (float*)(smem + 12288);         // [4][768]
  float*  s_target = (float*)(smem + 24576);         // [4][768]
  float*  s_cfeat  = (float*)(smem + 36864);         // [4][768]
  float*  s_tfeat  = (float*)(smem + 49152);         // [4][768]
  float*  s_ht     = (float*)(smem + 61440);         // [4][768] (also x2 temp)
  float*  s_red    = (float*)(smem + 73728);         // [64]
  bf16_t* s_xa     = (bf16_t*)(smem + 73984);        // [16][768]  A staging
  bf16_t* s_gateA  = (bf16_t*)(smem + 98560);        // [16][1536] concat(emb,core)
  bf16_t* s_big    = (bf16_t*)(smem + 147712);       // [16][3072] hier1 / t1

  const int tid   = threadIdx.x;
  const int lane  = tid & 31;
  const int wave  = tid >> 5;
  const int nthr  = blockDim.x;
  const int nwav  = nthr >> 5;

  // -- stage 0: gather embedding rows, L2-normalize ------------------------
  for (int i = tid; i < BATCH * D_MODEL; i += nthr) {
    int b = i / D_MODEL, d = i - b * D_MODEL;
    int tok = token_ids[b * SEQ_T + t];
    s_embN[i] = embedding[(size_t)tok * D_MODEL + d];
  }
  __syncthreads();
  if (wave < BATCH) {
    float s = 0.f;
    for (int d = lane; d < D_MODEL; d += 32) { float v = s_embN[wave * D_MODEL + d]; s += v * v; }
    s = wave_sum(s);
    if (lane == 0) s_red[wave] = 1.0f / fmaxf(sqrtf(s), 1e-12f);
  }
  __syncthreads();
  for (int i = tid; i < BATCH * D_MODEL; i += nthr) {
    int b = i / D_MODEL;
    float e = s_embN[i] * s_red[b];
    s_embN[i] = e;
    s_gateA[b * (2 * D_MODEL) + (i - b * D_MODEL)] = (bf16_t)e;   // gate A, 1st half
  }
  __syncthreads();

  // -- stage 1: temporal_raw = h_prev @ R  (24 groups of 2 tiles) ----------
  for (int grp = wave; grp < D_MODEL / 32; grp += nwav) {
    int nb = grp * 32;
    v8f acc[2];
    tile_gemm_nb<2>(h_state, D_MODEL, RT, D_MODEL, nb, lane, acc);
    if (lane < 16) {
      #pragma unroll
      for (int j = 0; j < 2; ++j)
        #pragma unroll
        for (int r = 0; r < BATCH; ++r) {
          int n = nb + j * 16 + lane;
          s_traw[r * D_MODEL + n] = acc[j][r];
          s_xa[r * D_MODEL + n]   = (bf16_t)acc[j][r];
        }
    }
  }
  __syncthreads();

  // -- stage 1b: temporal_feat = temporal_raw @ Wt -------------------------
  for (int grp = wave; grp < D_MODEL / 32; grp += nwav) {
    int nb = grp * 32;
    v8f acc[2];
    tile_gemm_nb<2>(s_xa, D_MODEL, WtT, D_MODEL, nb, lane, acc);
    if (lane < 16)
      #pragma unroll
      for (int j = 0; j < 2; ++j)
        #pragma unroll
        for (int r = 0; r < BATCH; ++r)
          s_tfeat[r * D_MODEL + nb + j * 16 + lane] = acc[j][r];
  }
  __syncthreads();

  // -- stage 2: ctx = emb + alpha*traw  (bf16 A in s_xa) -------------------
  for (int i = tid; i < BATCH * D_MODEL; i += nthr)
    s_xa[i] = (bf16_t)(s_embN[i] + ALPHA * s_traw[i]);
  __syncthreads();

  // -- stage 3: hier1 = gelu(ctx @ V0 + b0)  (96 groups, N=3072) -----------
  for (int grp = wave; grp < H_DIM / 32; grp += nwav) {
    int nb = grp * 32;
    v8f acc[2];
    tile_gemm_nb<2>(s_xa, D_MODEL, V0T, D_MODEL, nb, lane, acc);
    if (lane < 16)
      #pragma unroll
      for (int j = 0; j < 2; ++j)
        #pragma unroll
        for (int r = 0; r < BATCH; ++r) {
          int n = nb + j * 16 + lane;
          s_big[r * H_DIM + n] = (bf16_t)gelu_exact(acc[j][r] + b0v[n]);
        }
  }
  __syncthreads();

  // -- stage 4: target = gelu(hier1 @ V1 + b1)  (K=3072) -------------------
  for (int grp = wave; grp < D_MODEL / 32; grp += nwav) {
    int nb = grp * 32;
    v8f acc[2];
    tile_gemm_nb<2>(s_big, H_DIM, V1T, H_DIM, nb, lane, acc);
    if (lane < 16)
      #pragma unroll
      for (int j = 0; j < 2; ++j)
        #pragma unroll
        for (int r = 0; r < BATCH; ++r) {
          int n = nb + j * 16 + lane;
          s_target[r * D_MODEL + n] = gelu_exact(acc[j][r] + b1v[n]);
        }
  }
  __syncthreads();

  // -- stage 5: x2 relax (10 halvings == lerp by 2^-10), L2 norms, blend ---
  for (int i = tid; i < BATCH * D_MODEL; i += nthr) {
    float tg = s_target[i];
    float xn = tg + (x2_state[i] - tg) * 0.0009765625f;   // 2^-10
    x2_state[i] = xn;
    s_ht[i] = xn;                                          // temp
  }
  __syncthreads();
  if (wave < BATCH) {
    float s = 0.f;
    for (int d = lane; d < D_MODEL; d += 32) { float v = s_target[wave * D_MODEL + d]; s += v * v; }
    s = wave_sum(s);
    if (lane == 0) s_red[wave] = 1.0f / fmaxf(sqrtf(s), 1e-12f);
  } else if (wave < 2 * BATCH) {
    int row = wave - BATCH;
    float s = 0.f;
    for (int d = lane; d < D_MODEL; d += 32) { float v = s_ht[row * D_MODEL + d]; s += v * v; }
    s = wave_sum(s);
    if (lane == 0) s_red[8 + row] = 1.0f / fmaxf(sqrtf(s), 1e-12f);
  }
  __syncthreads();
  for (int i = tid; i < BATCH * D_MODEL; i += nthr) {
    int b = i / D_MODEL;
    s_xa[i] = (bf16_t)(0.5f * (s_target[i] * s_red[b] + s_ht[i] * s_red[8 + b]));
  }
  __syncthreads();

  // -- stage 6: t1 = gelu(blended @ W1 + c1) -------------------------------
  for (int grp = wave; grp < D_MODEL / 32; grp += nwav) {
    int nb = grp * 32;
    v8f acc[2];
    tile_gemm_nb<2>(s_xa, D_MODEL, W1T, D_MODEL, nb, lane, acc);
    if (lane < 16)
      #pragma unroll
      for (int j = 0; j < 2; ++j)
        #pragma unroll
        for (int r = 0; r < BATCH; ++r) {
          int n = nb + j * 16 + lane;
          s_big[r * D_MODEL + n] = (bf16_t)gelu_exact(acc[j][r] + c1v[n]);
        }
  }
  __syncthreads();

  // -- stage 7: core_feat = t1 @ W2 + c2 -----------------------------------
  for (int grp = wave; grp < D_MODEL / 32; grp += nwav) {
    int nb = grp * 32;
    v8f acc[2];
    tile_gemm_nb<2>(s_big, D_MODEL, W2T, D_MODEL, nb, lane, acc);
    if (lane < 16)
      #pragma unroll
      for (int j = 0; j < 2; ++j)
        #pragma unroll
        for (int r = 0; r < BATCH; ++r) {
          int n = nb + j * 16 + lane;
          float cf = acc[j][r] + c2v[n];
          s_cfeat[r * D_MODEL + n] = cf;
          s_gateA[r * (2 * D_MODEL) + D_MODEL + n] = (bf16_t)cf;  // gate A, 2nd half
        }
  }
  __syncthreads();

  // -- stage 8: gate = sigmoid([emb,core] @ Wg + bg); blend into h ---------
  for (int grp = wave; grp < D_MODEL / 32; grp += nwav) {
    int nb = grp * 32;
    v8f acc[2];
    tile_gemm_nb<2>(s_gateA, 2 * D_MODEL, WgT, 2 * D_MODEL, nb, lane, acc);
    if (lane < 16) {
      #pragma unroll
      for (int j = 0; j < 2; ++j) {
        int n = nb + j * 16 + lane;
        #pragma unroll
        for (int r = 0; r < BATCH; ++r) {
          float g  = 1.0f / (1.0f + expf(-(acc[j][r] + bgv[n])));
          float hv = g * (s_cfeat[r * D_MODEL + n] + ALPHA * s_tfeat[r * D_MODEL + n])
                   + (1.0f - g) * s_embN[r * D_MODEL + n];
          s_ht[r * D_MODEL + n] = hv;
        }
      }
    }
  }
  __syncthreads();

  // -- stage 9: LayerNorm, write h_t (bf16) for lm_head + next step --------
  if (wave < BATCH) {
    int row = wave;
    float s = 0.f;
    for (int d = lane; d < D_MODEL; d += 32) s += s_ht[row * D_MODEL + d];
    float mean = wave_sum(s) * (1.0f / D_MODEL);
    float v = 0.f;
    for (int d = lane; d < D_MODEL; d += 32) {
      float df = s_ht[row * D_MODEL + d] - mean; v += df * df;
    }
    float rs = rsqrtf(wave_sum(v) * (1.0f / D_MODEL) + 1e-5f);
    for (int d = lane; d < D_MODEL; d += 32) {
      float hn = (s_ht[row * D_MODEL + d] - mean) * rs * gamv[d] + betv[d];
      h_state[row * D_MODEL + d] = (bf16_t)hn;
    }
  }
}

// ---------------------------------------------------------------------------
// lm_head: logits[b,t,:] = h_t @ Wl.  h (24KB) staged in LDS once per block;
// each wave computes 4 adjacent 16-wide column tiles (1 A ds_load + 4 B
// global loads + 4 WMMAs per K-step).  WlT (bf16, 77MB) stays L2-resident.
// ---------------------------------------------------------------------------
__global__ void __launch_bounds__(256)
lm_head_kernel(const bf16_t* __restrict__ h, const bf16_t* __restrict__ WlT,
               float* __restrict__ out, int t) {
  __shared__ bf16_t sh[16 * D_MODEL];
  for (int i = threadIdx.x; i < (16 * D_MODEL) / 8; i += blockDim.x)
    ((uint4*)sh)[i] = ((const uint4*)h)[i];     // 16B chunks: global -> LDS
  __syncthreads();

  const int lane = threadIdx.x & 31;
  const int wave = threadIdx.x >> 5;
  const int grp  = blockIdx.x * (blockDim.x >> 5) + wave;
  const int ngrp = VOCAB_PAD / 64;              // 786
  if (grp >= ngrp) return;                      // wave-uniform
  const int nb0 = grp * 64;

  v8f acc[4];
  tile_gemm_nb<4>(sh, D_MODEL, WlT, D_MODEL, nb0, lane, acc);

  if (lane < 16) {
    #pragma unroll
    for (int j = 0; j < 4; ++j) {
      int v = nb0 + j * 16 + lane;
      if (v < VOCAB) {
        #pragma unroll
        for (int r = 0; r < BATCH; ++r)
          out[((size_t)r * SEQ_T + t) * VOCAB + v] = acc[j][r];
      }
    }
  }
}

// ---------------------------------------------------------------------------
// host driver
// ---------------------------------------------------------------------------
extern "C" void kernel_launch(void* const* d_in, const int* in_sizes, int n_in,
                              void* d_out, int out_size, void* d_ws, size_t ws_size,
                              hipStream_t stream) {
  const int*   token_ids = (const int*)d_in[0];
  const float* embedding = (const float*)d_in[1];
  const float* V0 = (const float*)d_in[2];  const float* b0 = (const float*)d_in[3];
  const float* V1 = (const float*)d_in[4];  const float* b1 = (const float*)d_in[5];
  const float* W1 = (const float*)d_in[6];  const float* c1 = (const float*)d_in[7];
  const float* W2 = (const float*)d_in[8];  const float* c2 = (const float*)d_in[9];
  const float* Wg = (const float*)d_in[10]; const float* bg = (const float*)d_in[11];
  const float* Wt = (const float*)d_in[12]; const float* gm = (const float*)d_in[13];
  const float* bt = (const float*)d_in[14]; const float* Wl = (const float*)d_in[15];
  const float* Rw = (const float*)d_in[16];

  char* p = (char*)d_ws;
  bf16_t* WlT = (bf16_t*)p; p += (size_t)VOCAB_PAD * D_MODEL * 2;
  bf16_t* V0T = (bf16_t*)p; p += (size_t)H_DIM * D_MODEL * 2;
  bf16_t* V1T = (bf16_t*)p; p += (size_t)D_MODEL * H_DIM * 2;
  bf16_t* W1T = (bf16_t*)p; p += (size_t)D_MODEL * D_MODEL * 2;
  bf16_t* W2T = (bf16_t*)p; p += (size_t)D_MODEL * D_MODEL * 2;
  bf16_t* WgT = (bf16_t*)p; p += (size_t)D_MODEL * 2 * D_MODEL * 2;
  bf16_t* WtT = (bf16_t*)p; p += (size_t)D_MODEL * D_MODEL * 2;
  bf16_t* RT  = (bf16_t*)p; p += (size_t)D_MODEL * D_MODEL * 2;
  bf16_t* h   = (bf16_t*)p; p += (size_t)16 * D_MODEL * 2;
  float*  x2  = (float*)p;  p += (size_t)BATCH * D_MODEL * 4;

  auto launch_T = [&](const float* src, bf16_t* dst, int K, int N, int Npad) {
    size_t total = (size_t)Npad * K;
    int blocks = (int)((total + 255) / 256);
    convert_transpose<<<blocks, 256, 0, stream>>>(src, dst, K, N, Npad);
  };
  launch_T(Wl, WlT, D_MODEL, VOCAB, VOCAB_PAD);
  launch_T(V0, V0T, D_MODEL, H_DIM, H_DIM);
  launch_T(V1, V1T, H_DIM, D_MODEL, D_MODEL);
  launch_T(W1, W1T, D_MODEL, D_MODEL, D_MODEL);
  launch_T(W2, W2T, D_MODEL, D_MODEL, D_MODEL);
  launch_T(Wg, WgT, 2 * D_MODEL, D_MODEL, D_MODEL);
  launch_T(Wt, WtT, D_MODEL, D_MODEL, D_MODEL);
  launch_T(Rw, RT,  D_MODEL, D_MODEL, D_MODEL);
  init_state<<<(16 * D_MODEL + 255) / 256, 256, 0, stream>>>(h, x2);

  const size_t smem_bytes = 246016;             // ~240 KB LDS arena (<320 KB/WGP)
  const int lm_groups = VOCAB_PAD / 64;         // 786 (4-tile groups)
  const int lm_blocks = (lm_groups + 7) / 8;    // 8 waves / 256-thread block

  for (int t = 0; t < SEQ_T; ++t) {
    step_kernel<<<1, 512, smem_bytes, stream>>>(
        token_ids, embedding, b0, b1, c1, c2, bg, gm, bt,
        V0T, V1T, W1T, W2T, WgT, WtT, RT, h, x2, t);
    lm_head_kernel<<<lm_blocks, 256, 0, stream>>>(h, WlT, (float*)d_out, t);
  }
  (void)in_sizes; (void)n_in; (void)out_size; (void)ws_size;
}
